// KNN_43061342110166
// MI455X (gfx1250) — matrix operand order repaired and checked by reading, not compiled, not run on hardware
//
#include <hip/hip_runtime.h>
#include <hip/hip_bf16.h>

// ---------------------------------------------------------------------------
// KNN k-th distance on MI455X (gfx1250, wave32, WMMA).
//   dist_k = sqrt(2 - 2*dot_{(k+1)-th largest})  for unit-normalized rows.
// Pipeline: normalize->f16, WMMA GEMM + per-chunk exact top-11 (LDS),
//           final merge of chunk candidates.
// ---------------------------------------------------------------------------

typedef __attribute__((ext_vector_type(16))) _Float16 v16h;
typedef __attribute__((ext_vector_type(8)))  _Float16 v8h;
typedef __attribute__((ext_vector_type(4)))  _Float16 v4h;
typedef __attribute__((ext_vector_type(8)))  float    v8f;

#define N_Q      2048
#define M_REF    50000
#define DIM      128
#define KSEL     11                      // k = 10 -> need the 11 largest dots
#define CHUNK    960                     // columns per workgroup (LDS tile)
#define NCHUNK   ((M_REF + CHUNK - 1) / CHUNK)   // 53
#define GRP_PER_WAVE (CHUNK / 16 / 4)    // 15 col-groups of 16 per wave

// ---------------- workspace layout (bytes) ----------------
#define WS_ZN    0u
#define WS_RN    (N_Q * DIM * 2u)                       // 524288
#define WS_CAND  (WS_RN + (unsigned)(M_REF * DIM * 2))  // 13324288
// cand: [N_Q][NCHUNK][KSEL] float

// ---------------------------------------------------------------------------
// Kernel 1: row L2-normalize fp32 -> f16.  One wave per row (4 floats/lane).
// ---------------------------------------------------------------------------
__global__ void knn_normalize_f16(const float* __restrict__ in,
                                  _Float16* __restrict__ out, int rows) {
    int gid  = blockIdx.x * blockDim.x + threadIdx.x;
    int row  = gid >> 5;
    int lane = gid & 31;
    if (row >= rows) return;

    const float4 v = ((const float4*)(in + (size_t)row * DIM))[lane];
    float ss = v.x * v.x + v.y * v.y + v.z * v.z + v.w * v.w;
#pragma unroll
    for (int off = 16; off; off >>= 1) ss += __shfl_xor(ss, off, 32);
    float s = rsqrtf(ss);

    v4h o;
    o[0] = (_Float16)(v.x * s);
    o[1] = (_Float16)(v.y * s);
    o[2] = (_Float16)(v.z * s);
    o[3] = (_Float16)(v.w * s);
    ((v4h*)(out + (size_t)row * DIM))[lane] = o;
}

// ---------------------------------------------------------------------------
// Kernel 2: WMMA GEMM tile (16 queries x CHUNK refs, K=128) + exact per-row
// top-11 of the chunk.  Block = 128 threads (4 waves).
// ---------------------------------------------------------------------------
__global__ __launch_bounds__(128)
void knn_dot_topk(const _Float16* __restrict__ zn,
                  const _Float16* __restrict__ rn,
                  float* __restrict__ cand) {
    __shared__ float tile[16 * CHUNK];   // 60 KB

    const int qt   = blockIdx.x;         // query tile 0..127
    const int ch   = blockIdx.y;         // column chunk 0..NCHUNK-1
    const int tid  = threadIdx.x;
    const int wid  = tid >> 5;
    const int lane = tid & 31;
    const int half = lane >> 4;          // 0 | 1
    const int l15  = lane & 15;

    // ---- A fragments: 16 zn rows, K=128 in 4 steps of 32 ----------------
    // ISA 16-bit A 16x32 layout: lane(l15)=row M; halves 0-7 = K[8*half..],
    // halves 8-15 = K[16+8*half..]  => two contiguous 16B loads per lane.
    const _Float16* arow = zn + (size_t)(qt * 16 + l15) * DIM;
    union { v16h v; v8h h[2]; } A[4];
#pragma unroll
    for (int kk = 0; kk < 4; ++kk) {
        A[kk].h[0] = *(const v8h*)(arow + 32 * kk + 8 * half);
        A[kk].h[1] = *(const v8h*)(arow + 32 * kk + 16 + 8 * half);
    }

    // ---- GEMM: each wave covers GRP_PER_WAVE groups of 16 columns -------
    const int col0_wave = ch * CHUNK + wid * (GRP_PER_WAVE * 16);
#pragma unroll 1
    for (int g = 0; g < GRP_PER_WAVE; ++g) {
        const int colbase = col0_wave + g * 16;
        const int col     = colbase + l15;
        const int colc    = (col < M_REF) ? col : (M_REF - 1);
        const _Float16* brow = rn + (size_t)colc * DIM;

        v8f acc = {};
#pragma unroll
        for (int kk = 0; kk < 4; ++kk) {
            // ISA 16-bit B 32x16 layout: lane(l15)=col N; halves 0-7 = K[8*half..],
            // halves 8-15 = K[16+8*half..]  => two contiguous 16B loads per lane.
            v8h B0 = *(const v8h*)(brow + 32 * kk + 8 * half);
            v8h B1 = *(const v8h*)(brow + 32 * kk + 16 + 8 * half);
            v16h B = __builtin_shufflevector(B0, B1, 0, 1, 2, 3, 4, 5, 6, 7, 8, 9, 10, 11, 12, 13, 14, 15);
            acc = __builtin_amdgcn_wmma_f32_16x16x32_f16(
                false, A[kk].v, false, B, (short)0, acc, false, false);
        }

        // Scatter D to LDS: lane holds rows m = vg + 8*half of column l15.
        const int  lcol  = (colbase - ch * CHUNK) + l15;
        const bool valid = (col < M_REF);
#pragma unroll
        for (int vg = 0; vg < 8; ++vg) {
            int m = vg + 8 * half;
            tile[m * CHUNK + lcol] = valid ? acc[vg] : -2.0f;  // dot >= -1
        }
    }
    __syncthreads();

    // ---- Exact top-11 per row of the chunk: wave handles 4 rows ---------
#pragma unroll 1
    for (int rr = 0; rr < 4; ++rr) {
        const int m = wid * 4 + rr;
        float lst[KSEL];
#pragma unroll
        for (int i = 0; i < KSEL; ++i) lst[i] = -3.0f;

        const float* rowp = &tile[m * CHUNK];
#pragma unroll
        for (int i = 0; i < CHUNK / 32; ++i) {       // 30 values/lane
            float x = rowp[lane + 32 * i];
#pragma unroll
            for (int j = 0; j < KSEL; ++j) {         // sorted-desc insert
                float mx = fmaxf(lst[j], x);
                x        = fminf(lst[j], x);
                lst[j]   = mx;
            }
        }

        // Merge lanes: pop the wave-global max 11 times.
        const int row_global = qt * 16 + m;
        float* outp = cand + ((size_t)row_global * NCHUNK + ch) * KSEL;
#pragma unroll 1
        for (int it = 0; it < KSEL; ++it) {
            float v  = lst[0];
            int   idx = lane;
#pragma unroll
            for (int off = 16; off; off >>= 1) {
                float ov = __shfl_xor(v, off, 32);
                int   oi = __shfl_xor(idx, off, 32);
                if (ov > v || (ov == v && oi < idx)) { v = ov; idx = oi; }
            }
            if (lane == 0) outp[it] = v;
            const bool win = (idx == lane);
#pragma unroll
            for (int j = 0; j < KSEL - 1; ++j)
                lst[j] = win ? lst[j + 1] : lst[j];
            lst[KSEL - 1] = win ? -3.0f : lst[KSEL - 1];
        }
    }
}

// ---------------------------------------------------------------------------
// Kernel 3: merge NCHUNK*KSEL candidates per row, take (k+1)-th largest dot,
// emit sqrt(max(2 - 2*dot, 1e-12)).  One wave per row.
// ---------------------------------------------------------------------------
__global__ void knn_merge_final(const float* __restrict__ cand,
                                const int* __restrict__ kp,
                                float* __restrict__ out) {
    int gid  = blockIdx.x * blockDim.x + threadIdx.x;
    int row  = gid >> 5;
    int lane = gid & 31;
    if (row >= N_Q) return;

    int need = kp[0] + 1;                 // k-th smallest dist (0-based sort)
    if (need > KSEL) need = KSEL;
    if (need < 1)    need = 1;

    const float* p = cand + (size_t)row * (NCHUNK * KSEL);
    float lst[KSEL];
#pragma unroll
    for (int i = 0; i < KSEL; ++i) lst[i] = -3.0f;

    const int TOT = NCHUNK * KSEL;        // 583
    for (int i = lane; i < TOT; i += 32) {
        float x = p[i];
#pragma unroll
        for (int j = 0; j < KSEL; ++j) {
            float mx = fmaxf(lst[j], x);
            x        = fminf(lst[j], x);
            lst[j]   = mx;
        }
    }

    float kth = -3.0f;
    for (int it = 0; it < need; ++it) {
        float v  = lst[0];
        int   idx = lane;
#pragma unroll
        for (int off = 16; off; off >>= 1) {
            float ov = __shfl_xor(v, off, 32);
            int   oi = __shfl_xor(idx, off, 32);
            if (ov > v || (ov == v && oi < idx)) { v = ov; idx = oi; }
        }
        kth = v;
        const bool win = (idx == lane);
#pragma unroll
        for (int j = 0; j < KSEL - 1; ++j)
            lst[j] = win ? lst[j + 1] : lst[j];
        lst[KSEL - 1] = win ? -3.0f : lst[KSEL - 1];
    }

    if (lane == 0)
        out[row] = sqrtf(fmaxf(2.0f - 2.0f * kth, 1e-12f));
}

// ---------------------------------------------------------------------------
extern "C" void kernel_launch(void* const* d_in, const int* in_sizes, int n_in,
                              void* d_out, int out_size, void* d_ws, size_t ws_size,
                              hipStream_t stream) {
    const float* z   = (const float*)d_in[0];   // [2048,128] f32
    const float* ref = (const float*)d_in[1];   // [50000,128] f32
    const int*   kp  = (const int*)d_in[2];     // scalar k (=10)

    char* ws = (char*)d_ws;
    _Float16* zn   = (_Float16*)(ws + WS_ZN);
    _Float16* rn   = (_Float16*)(ws + WS_RN);
    float*    cand = (float*)   (ws + WS_CAND);
    float*    out  = (float*)d_out;             // [2048] f32

    // 1) normalize + f16 convert (one wave per row)
    {
        int thr = N_Q * 32;
        knn_normalize_f16<<<(thr + 255) / 256, 256, 0, stream>>>(z, zn, N_Q);
    }
    {
        int thr = M_REF * 32;
        knn_normalize_f16<<<(thr + 255) / 256, 256, 0, stream>>>(ref, rn, M_REF);
    }

    // 2) WMMA GEMM + per-chunk top-11
    {
        dim3 grid(N_Q / 16, NCHUNK);    // 128 x 53
        knn_dot_topk<<<grid, 128, 0, stream>>>(zn, rn, cand);
    }

    // 3) merge candidates -> k-th distance
    {
        int thr = N_Q * 32;
        knn_merge_final<<<(thr + 255) / 256, 256, 0, stream>>>(cand, kp, out);
    }
}
